// LSTM2_52982716563923
// MI455X (gfx1250) — compile-verified, matrix operand order checked by reading, not run on hardware
//
#include <hip/hip_runtime.h>
#include <hip/hip_bf16.h>
#include <stdint.h>

typedef __attribute__((ext_vector_type(16))) __bf16 v16bf;
typedef __attribute__((ext_vector_type(8)))  __bf16 v8bf;
typedef __attribute__((ext_vector_type(4)))  __bf16 v4bf;
typedef __attribute__((ext_vector_type(8)))  float  v8f;

constexpr int T_ = 512, B_ = 64, E_ = 512, H_ = 1024, O_ = 512;
constexpr int G_ = 4 * H_;          // 4096
constexpr int K_ = E_ + H_;         // 1536 (fused K for gates GEMM)
constexpr int APAD_ = 8;            // LDS row pad (bf16 elems) to spread banks
constexpr int AROW_ = K_ + APAD_;   // 1544
constexpr int HROW_ = H_ + APAD_;   // 1032

__device__ __forceinline__ float sigmoidf_(float x) {
    return 1.0f / (1.0f + __expf(-x));
}

__device__ __forceinline__ v16bf cat16_(v8bf lo, v8bf hi) {
    return __builtin_shufflevector(lo, hi, 0,1,2,3,4,5,6,7,8,9,10,11,12,13,14,15);
}

// Async 16-byte global -> LDS copy (no VGPR data path, tracked by ASYNCcnt).
__device__ __forceinline__ void async_copy_b128_(unsigned lds_off,
                                                 const void* gptr) {
    asm volatile("global_load_async_to_lds_b128 %0, %1, off"
                 :
                 : "v"(lds_off), "v"((uint64_t)(uintptr_t)gptr)
                 : "memory");
}
__device__ __forceinline__ void wait_asynccnt0_() {
    asm volatile("s_wait_asynccnt 0" ::: "memory");
}

// ---------------- fp32 -> bf16 conversion (4 elems / thread) ----------------
__global__ void f32_to_bf16_kernel(const float* __restrict__ src,
                                   __bf16* __restrict__ dst, int n) {
    int i = (blockIdx.x * blockDim.x + threadIdx.x) * 4;
    if (i >= n) return;
    float4 v = *(const float4*)(src + i);
    v4bf o;
    o[0] = (__bf16)v.x; o[1] = (__bf16)v.y; o[2] = (__bf16)v.z; o[3] = (__bf16)v.w;
    *(v4bf*)(dst + i) = o;
}

// ---------------- pack W [K,N] row-major fp32 into WMMA B-fragment order ----
// dst[((kt*tileN + nt)*32 + lane)*16 + e] = (bf16) W[k, n]
//   n = nt*16 + (lane & 15)
//   kbase = (lane>>4)*8 ; k = kt*32 + ((e<8) ? kbase+e : 16+kbase+(e-8))
__global__ void pack_weight_kernel(const float* __restrict__ W, int K, int N,
                                   __bf16* __restrict__ dst) {
    int idx = blockIdx.x * blockDim.x + threadIdx.x;
    if (idx >= K * N) return;
    int e    = idx & 15;
    int lane = (idx >> 4) & 31;
    int tid  = idx >> 9;            // tile index = kt*tileN + nt
    int tileN = N >> 4;
    int nt = tid % tileN;
    int kt = tid / tileN;
    int col = lane & 15;
    int kb  = (lane >> 4) * 8;
    int kl  = (e < 8) ? (kb + e) : (16 + kb + (e - 8));
    int k   = kt * 32 + kl;
    int n   = nt * 16 + col;
    dst[idx] = (__bf16)W[(size_t)k * N + n];
}

// ---------------- init h, c (reference sets c0 = h0) ------------------------
__global__ void init_hc_kernel(const float* __restrict__ h0,
                               float* __restrict__ h, float* __restrict__ c,
                               __bf16* __restrict__ hb) {
    int i = blockIdx.x * blockDim.x + threadIdx.x;
    if (i >= B_ * H_) return;
    float v = h0[i];
    h[i] = v;
    c[i] = v;
    hb[i] = (__bf16)v;
}

// ---------------- gates = [x_t | h] @ [Wi ; Wh] + bi  (K = 1536) ------------
// Grid: 32 blocks x 256 threads. Block = (mt, ng): mt = M-tile (16 rows of B),
// ng = group of 32 N-tiles. Each block stages its 16x1536 A panel in LDS via
// async global->LDS b128 copies; each wave accumulates 4 N-tiles per A
// fragment (4 WMMA per A load from LDS).
__global__ __launch_bounds__(256) void lstm_gates_kernel(
    const __bf16* __restrict__ xt,   // [B,E] bf16 (this timestep)
    const __bf16* __restrict__ hb,   // [B,H] bf16
    const __bf16* __restrict__ Wih,  // packed [48 ktiles][256 ntiles][32][16]
    const float*  __restrict__ bi,   // [G]
    float*        __restrict__ gates)// [B,G] fp32
{
    __shared__ __bf16 sA[16 * AROW_];   // 49,408 bytes

    const int tid  = threadIdx.x;
    const int lane = tid & 31;
    const int wave = tid >> 5;
    const int mt = blockIdx.x >> 3;     // 0..3
    const int ng = blockIdx.x & 7;      // 0..7

    // Async stage: 16 rows x 1536 bf16 = 3072 x 16B chunks, 12 / thread.
    // Low 32 bits of the generic shared pointer == raw LDS byte offset.
    const unsigned sA_base = (unsigned)(uintptr_t)(void*)sA;
    for (int ch = tid; ch < 16 * (K_ / 8); ch += 256) {
        int r  = ch / (K_ / 8);         // 0..15
        int c8 = ch % (K_ / 8);         // 0..191
        int row = mt * 16 + r;
        const void* g = (c8 < E_ / 8)
            ? (const void*)(xt + (size_t)row * E_ + c8 * 8)
            : (const void*)(hb + (size_t)row * H_ + (c8 - E_ / 8) * 8);
        async_copy_b128_(sA_base + (unsigned)(r * AROW_ + c8 * 8) * 2, g);
    }
    wait_asynccnt0_();
    __syncthreads();

    const int kb = (lane >> 4) * 8;
    const __bf16* arow = sA + (lane & 15) * AROW_;
    const int nt0 = ng * 32 + wave * 4;         // 4 consecutive N-tiles

    v8f acc[4] = {};
    for (int kt = 0; kt < 48; ++kt) {
        v8bf alo = *(const v8bf*)(arow + kt * 32 + kb);
        v8bf ahi = *(const v8bf*)(arow + kt * 32 + 16 + kb);
        v16bf av = cat16_(alo, ahi);
        const __bf16* bbase = Wih + (((size_t)(kt * 256 + nt0) * 32 + lane) << 4);
        #pragma unroll
        for (int j = 0; j < 4; ++j) {
            v8bf blo = *(const v8bf*)(bbase + j * 512);
            v8bf bhi = *(const v8bf*)(bbase + j * 512 + 8);
            v16bf bv = cat16_(blo, bhi);
            acc[j] = __builtin_amdgcn_wmma_f32_16x16x32_bf16(
                false, av, false, bv, (short)0, acc[j], false, false);
        }
    }

    // D layout: lanes 0-15 -> M=r, lanes 16-31 -> M=r+8; N = lane&15
    const int m0  = (lane >> 4) * 8;
    const int col = lane & 15;
    #pragma unroll
    for (int j = 0; j < 4; ++j) {
        const int n = (nt0 + j) * 16 + col;
        const float bias = bi[n];
        #pragma unroll
        for (int r = 0; r < 8; ++r)
            gates[(size_t)(mt * 16 + m0 + r) * G_ + n] = acc[j][r] + bias;
    }
}

// ---------------- fused cell update + out_t = h @ Wo + bo -------------------
// Grid: 4 blocks x 256 threads; block mt owns rows [mt*16, mt*16+16).
// Phase 1: elementwise LSTM update for its rows; h (bf16) -> LDS + global.
// Phase 2: 8 waves x 4 N-tiles = 32 N-tiles = full O = 512 columns.
__global__ __launch_bounds__(256) void lstm_update_out_kernel(
    const float*  __restrict__ gates, // [B,G] fp32
    float*        __restrict__ c,     // [B,H] fp32 (state)
    float*        __restrict__ h,     // [B,H] fp32 (state, for final output)
    __bf16*       __restrict__ hbg,   // [B,H] bf16 (for next step's gates GEMM)
    const __bf16* __restrict__ Wop,   // packed [32 ktiles][32 ntiles][32][16]
    const float*  __restrict__ bo,    // [O]
    float*        __restrict__ out)   // [B,O] fp32 (this timestep's slice)
{
    __shared__ __bf16 sH[16 * HROW_];   // 33,024 bytes

    const int tid = threadIdx.x;
    const int mt  = blockIdx.x;         // 0..3

    // Phase 1: LSTM cell update for 16 rows (16*1024 elems, 64 / thread)
    for (int idx = tid; idx < 16 * H_; idx += 256) {
        int r = idx >> 10;              // /H
        int j = idx & (H_ - 1);
        int b = mt * 16 + r;
        const float* g = gates + (size_t)b * G_;
        float ig = sigmoidf_(g[j]);
        float fg = sigmoidf_(g[j + H_]);
        float gg = tanhf(g[j + 2 * H_]);
        float og = sigmoidf_(g[j + 3 * H_]);
        size_t ci = (size_t)b * H_ + j;
        float cn = c[ci] * fg + ig * gg;
        float hn = og * tanhf(cn);
        c[ci] = cn;
        h[ci] = hn;
        __bf16 hv = (__bf16)hn;
        hbg[ci] = hv;
        sH[r * HROW_ + j] = hv;
    }
    __syncthreads();

    // Phase 2: 16x512 output tile, K = 1024 (32 ktiles), A from LDS
    const int lane = tid & 31;
    const int wave = tid >> 5;
    const int kb = (lane >> 4) * 8;
    const __bf16* arow = sH + (lane & 15) * HROW_;
    const int nt0 = wave * 4;

    v8f acc[4] = {};
    for (int kt = 0; kt < 32; ++kt) {
        v8bf alo = *(const v8bf*)(arow + kt * 32 + kb);
        v8bf ahi = *(const v8bf*)(arow + kt * 32 + 16 + kb);
        v16bf av = cat16_(alo, ahi);
        const __bf16* bbase = Wop + (((size_t)(kt * 32 + nt0) * 32 + lane) << 4);
        #pragma unroll
        for (int j = 0; j < 4; ++j) {
            v8bf blo = *(const v8bf*)(bbase + j * 512);
            v8bf bhi = *(const v8bf*)(bbase + j * 512 + 8);
            v16bf bv = cat16_(blo, bhi);
            acc[j] = __builtin_amdgcn_wmma_f32_16x16x32_bf16(
                false, av, false, bv, (short)0, acc[j], false, false);
        }
    }

    const int m0  = (lane >> 4) * 8;
    const int col = lane & 15;
    #pragma unroll
    for (int j = 0; j < 4; ++j) {
        const int n = (nt0 + j) * 16 + col;
        const float bias = bo[n];
        #pragma unroll
        for (int r = 0; r < 8; ++r)
            out[(size_t)(mt * 16 + m0 + r) * O_ + n] = acc[j][r] + bias;
    }
}

// ---------------- final h_t copy into output tail ---------------------------
__global__ void final_h_kernel(const float* __restrict__ h, float* __restrict__ dst) {
    int i = blockIdx.x * blockDim.x + threadIdx.x;
    if (i >= B_ * H_) return;
    dst[i] = h[i];
}

extern "C" void kernel_launch(void* const* d_in, const int* in_sizes, int n_in,
                              void* d_out, int out_size, void* d_ws, size_t ws_size,
                              hipStream_t stream) {
    const float* x  = (const float*)d_in[0];   // [T,B,E]
    const float* h0 = (const float*)d_in[1];   // [B,H]
    const float* Wi = (const float*)d_in[2];   // [E,4H]
    const float* Wh = (const float*)d_in[3];   // [H,4H]
    const float* Wo = (const float*)d_in[4];   // [H,O]
    const float* bi = (const float*)d_in[5];   // [4H]
    const float* bo = (const float*)d_in[6];   // [O]
    float* out = (float*)d_out;                // [T,B,O] ++ [B,H]

    char* ws = (char*)d_ws;
    size_t off = 0;
    auto alloc = [&](size_t bytes) -> void* {
        void* p = ws + off;
        off += (bytes + 255) & ~(size_t)255;
        return p;
    };
    __bf16* xb    = (__bf16*)alloc((size_t)T_ * B_ * E_ * 2);        // 32 MB
    __bf16* Wih   = (__bf16*)alloc((size_t)K_ * G_ * 2);             // 12 MB
    __bf16* Wop   = (__bf16*)alloc((size_t)H_ * O_ * 2);             //  1 MB
    float*  gates = (float*) alloc((size_t)B_ * G_ * 4);             //  1 MB
    float*  hbuf  = (float*) alloc((size_t)B_ * H_ * 4);
    float*  cbuf  = (float*) alloc((size_t)B_ * H_ * 4);
    __bf16* hb    = (__bf16*)alloc((size_t)B_ * H_ * 2);

    // Preamble: conversions + weight packing + state init
    f32_to_bf16_kernel<<<(T_ * B_ * E_ / 4 + 255) / 256, 256, 0, stream>>>(
        x, xb, T_ * B_ * E_);
    pack_weight_kernel<<<(E_ * G_ + 255) / 256, 256, 0, stream>>>(Wi, E_, G_, Wih);
    pack_weight_kernel<<<(H_ * G_ + 255) / 256, 256, 0, stream>>>(
        Wh, H_, G_, Wih + (size_t)E_ * G_);
    pack_weight_kernel<<<(H_ * O_ + 255) / 256, 256, 0, stream>>>(Wo, H_, O_, Wop);
    init_hc_kernel<<<(B_ * H_ + 255) / 256, 256, 0, stream>>>(h0, hbuf, cbuf, hb);

    // Recurrence: 2 launches per step
    for (int t = 0; t < T_; ++t) {
        lstm_gates_kernel<<<32, 256, 0, stream>>>(
            xb + (size_t)t * B_ * E_, hb, Wih, bi, gates);
        lstm_update_out_kernel<<<4, 256, 0, stream>>>(
            gates, cbuf, hbuf, hb, Wop, bo, out + (size_t)t * B_ * O_);
    }

    final_h_kernel<<<(B_ * H_ + 255) / 256, 256, 0, stream>>>(
        hbuf, out + (size_t)T_ * B_ * O_);
}
